// LinearAttention_71116068487389
// MI455X (gfx1250) — compile-verified, matrix operand order checked by reading, not run on hardware
//
#include <hip/hip_runtime.h>
#include <hip/hip_bf16.h>
#include <math.h>
#include <stdint.h>

// ---- problem constants ----
#define BB   4
#define NN   4096
#define CC   1024
#define HH   16
#define HD   64
#define FF   3072            // 3*C
#define EPS  1e-6f

typedef __attribute__((ext_vector_type(16))) __bf16 v16bf;
typedef __attribute__((ext_vector_type(8)))  float  v8f;
typedef int v4i_g __attribute__((vector_size(16)));   // matches builtin param type

union Frag {
    v16bf v;
    uint4 u[2];
};

// ---- CDNA5 async global->LDS path (guarded; sync fallback otherwise) ----
#if defined(__has_builtin)
#  if __has_builtin(__builtin_amdgcn_global_load_async_to_lds_b128) && \
      __has_builtin(__builtin_amdgcn_s_wait_asynccnt)
#    define HAS_ASYNC 1
#  endif
#endif
#ifndef HAS_ASYNC
#  define HAS_ASYNC 0
#endif

static __device__ __forceinline__ void async_cp16(const __bf16* g, __bf16* l) {
#if HAS_ASYNC
    __builtin_amdgcn_global_load_async_to_lds_b128(
        (__attribute__((address_space(1))) v4i_g*)(uintptr_t)g,
        (__attribute__((address_space(3))) v4i_g*)(unsigned)(uintptr_t)l,
        0, 0);
#else
    *(uint4*)l = *(const uint4*)g;
#endif
}

#if HAS_ASYNC
#  define WAIT_ASYNC(n) __builtin_amdgcn_s_wait_asynccnt(n)
#else
#  define WAIT_ASYNC(n)
#endif

static __device__ __forceinline__ float silu(float x) {
    return x / (1.0f + __expf(-x));
}

// ------------------------------------------------------------------
// fp32 -> bf16 conversion (plain)
// ------------------------------------------------------------------
__global__ __launch_bounds__(256) void cvt_bf16_kernel(const float* __restrict__ src,
                                                       __bf16* __restrict__ dst, int n) {
    int i = blockIdx.x * 256 + threadIdx.x;
    if (i < n) dst[i] = (__bf16)src[i];
}

// ------------------------------------------------------------------
// fp32 -> bf16 transpose: src (R x Ccols) row-major -> dst (Ccols x R)
// ------------------------------------------------------------------
__global__ __launch_bounds__(256) void cvt_bf16_t_kernel(const float* __restrict__ src,
                                                         __bf16* __restrict__ dst,
                                                         int R, int Ccols) {
    int i = blockIdx.x * 256 + threadIdx.x;
    if (i < R * Ccols) {
        int r = i / Ccols;
        int c = i - r * Ccols;
        dst[(size_t)c * R + r] = (__bf16)src[i];
    }
}

// ==================================================================
// Shared GEMM tile machinery:
//   block = 256 thr (8 waves, 2m x 4n), tile 128(M) x 256(N), BK=32.
//   A (rows x 1024) row-major bf16; B supplied TRANSPOSED (cols x 1024)
//   row-major bf16 -> both tiles staged as contiguous 16B chunks via
//   async global->LDS copies, double buffered.
// ==================================================================
#define LDA 40   // padded k-stride in LDS (80B, 16B aligned)

static __device__ __forceinline__ void stage_tiles(const __bf16* __restrict__ A,
                                                   const __bf16* __restrict__ BT,
                                                   int row0, int col0, int kk,
                                                   __bf16* As, __bf16* Bs, int tid) {
    // A tile: 128 rows x 32 k = 512 chunks of 8 elems
    #pragma unroll
    for (int ii = 0; ii < 2; ++ii) {
        int ch = tid + ii * 256;
        int r  = ch >> 2;
        int s  = (ch & 3) * 8;
        async_cp16(A + (size_t)(row0 + r) * CC + kk + s, &As[r * LDA + s]);
    }
    // B tile: 256 cols x 32 k = 1024 chunks
    #pragma unroll
    for (int ii = 0; ii < 4; ++ii) {
        int ch = tid + ii * 256;
        int r  = ch >> 2;
        int s  = (ch & 3) * 8;
        async_cp16(BT + (size_t)(col0 + r) * CC + kk + s, &Bs[r * LDA + s]);
    }
}

// ------------------------------------------------------------------
// GEMM1: qkv = X(16384x1024) * Wqkv ; bias ; SiLU on q,k ;
// scatter to q (b,h,n,d), kT (b,h,d,n), vT (b,h,d,n) in bf16.
// ------------------------------------------------------------------
__global__ __launch_bounds__(256) void gemm_qkv_kernel(const __bf16* __restrict__ X,
                                                       const __bf16* __restrict__ WT,
                                                       const float*  __restrict__ bias,
                                                       __bf16* __restrict__ qs,
                                                       __bf16* __restrict__ kT,
                                                       __bf16* __restrict__ vT) {
    const int tid    = threadIdx.x;
    const int lane   = tid & 31;
    const int wid    = tid >> 5;
    const int wave_m = wid >> 2;          // 0..1  (64 rows)
    const int wave_n = wid & 3;           // 0..3  (64 cols)
    const int row0   = blockIdx.y * 128;
    const int col0   = blockIdx.x * 256;

    __shared__ __bf16 As[2][128 * LDA];
    __shared__ __bf16 Bs[2][256 * LDA];

    v8f acc[4][4] = {};
    const int ml = lane & 15;
    const int kb = (lane < 16) ? 0 : 8;
    const int NT = CC / 32;

    stage_tiles(X, WT, row0, col0, 0, As[0], Bs[0], tid);

    for (int it = 0; it < NT; ++it) {
        int cur = it & 1;
        if (it + 1 < NT) {
            stage_tiles(X, WT, row0, col0, (it + 1) * 32, As[1 - cur], Bs[1 - cur], tid);
            WAIT_ASYNC(6);
        } else {
            WAIT_ASYNC(0);
        }
        __syncthreads();

        const __bf16* Ab = As[cur];
        const __bf16* Bb = Bs[cur];

        // B fragments (shared across the 4 A-row tiles) stay live;
        // A fragments are streamed one at a time to cap VGPR pressure.
        Frag b[4];
        #pragma unroll
        for (int j = 0; j < 4; ++j) {
            int n = wave_n * 64 + j * 16 + ml;
            b[j].u[0] = *(const uint4*)&Bb[n * LDA + kb];
            b[j].u[1] = *(const uint4*)&Bb[n * LDA + kb + 16];
        }
        #pragma unroll
        for (int i = 0; i < 4; ++i) {
            Frag a;
            int m = wave_m * 64 + i * 16 + ml;
            a.u[0] = *(const uint4*)&Ab[m * LDA + kb];
            a.u[1] = *(const uint4*)&Ab[m * LDA + kb + 16];
            #pragma unroll
            for (int j = 0; j < 4; ++j)
                acc[i][j] = __builtin_amdgcn_wmma_f32_16x16x32_bf16(
                    false, a.v, false, b[j].v, (short)0, acc[i][j], false, false);
        }
        __syncthreads();
    }

    // epilogue: bias, silu(q,k), scatter
    #pragma unroll
    for (int i = 0; i < 4; ++i) {
        #pragma unroll
        for (int j = 0; j < 4; ++j) {
            #pragma unroll
            for (int e = 0; e < 8; ++e) {
                int mloc = wave_m * 64 + i * 16 + ((lane < 16) ? e : e + 8);
                int nloc = wave_n * 64 + j * 16 + ml;
                int r = row0 + mloc;                     // 0..16383
                int f = col0 + nloc;                     // 0..3071
                float val = acc[i][j][e] + bias[f];
                int bb = r >> 12;                        // / N
                int nn = r & (NN - 1);
                int h  = f / (3 * HD);
                int jj = f - h * (3 * HD);
                size_t head = (size_t)bb * HH + h;
                if (jj < HD) {
                    qs[(head * NN + nn) * HD + jj] = (__bf16)silu(val);
                } else if (jj < 2 * HD) {
                    kT[(head * HD + (jj - HD)) * NN + nn] = (__bf16)silu(val);
                } else {
                    vT[(head * HD + (jj - 2 * HD)) * NN + nn] = (__bf16)val;
                }
            }
        }
    }
}

// ------------------------------------------------------------------
// stateT[e][d] = sum_n v[n,e]*k[n,d]  (per b,h) + ksum[d] = sum_n k[n,d]
// ------------------------------------------------------------------
__global__ __launch_bounds__(128) void state_kernel(const __bf16* __restrict__ kT,
                                                    const __bf16* __restrict__ vT,
                                                    __bf16* __restrict__ stateT,
                                                    float*  __restrict__ ksum) {
    const int bh   = blockIdx.x;
    const int tid  = threadIdx.x;
    const int lane = tid & 31;
    const int w    = tid >> 5;
    const int ml   = lane & 15;
    const int kb   = (lane < 16) ? 0 : 8;

    const __bf16* kbase = kT + (size_t)bh * HD * NN;
    const __bf16* vbase = vT + (size_t)bh * HD * NN;

    v8f acc[4] = {};
    for (int kk = 0; kk < NN; kk += 32) {
        Frag a;
        const __bf16* ar = vbase + (size_t)(w * 16 + ml) * NN + kk + kb;
        a.u[0] = *(const uint4*)ar;
        a.u[1] = *(const uint4*)(ar + 16);
        #pragma unroll
        for (int j = 0; j < 4; ++j) {
            Frag b;
            const __bf16* br = kbase + (size_t)(j * 16 + ml) * NN + kk + kb;
            b.u[0] = *(const uint4*)br;
            b.u[1] = *(const uint4*)(br + 16);
            acc[j] = __builtin_amdgcn_wmma_f32_16x16x32_bf16(
                false, a.v, false, b.v, (short)0, acc[j], false, false);
        }
    }
    #pragma unroll
    for (int j = 0; j < 4; ++j)
        #pragma unroll
        for (int e = 0; e < 8; ++e) {
            int row = w * 16 + ((lane < 16) ? e : e + 8);   // e-feature
            int col = j * 16 + ml;                          // d-feature
            stateT[(size_t)bh * HD * HD + row * HD + col] = (__bf16)acc[j][e];
        }

    if (tid < HD) {
        const __bf16* rowp = kbase + (size_t)tid * NN;
        float s = 0.f;
        for (int i = 0; i < NN; i += 8) {
            union { uint4 u; __bf16 h[8]; } t;
            t.u = *(const uint4*)(rowp + i);
            #pragma unroll
            for (int q = 0; q < 8; ++q) s += (float)t.h[q];
        }
        ksum[bh * HD + tid] = s;
    }
}

// ------------------------------------------------------------------
// num = q @ state ; den = q . ksum ; attn[b,n,h*64+e] = num/(den+eps)
// ------------------------------------------------------------------
__global__ __launch_bounds__(128) void attn_kernel(const __bf16* __restrict__ qs,
                                                   const __bf16* __restrict__ stateT,
                                                   const float*  __restrict__ ksum,
                                                   __bf16* __restrict__ attn) {
    const int mtile = blockIdx.x;      // 0..63
    const int bh    = blockIdx.y;      // 0..63
    const int b     = bh >> 4;
    const int h     = bh & 15;
    const int tid   = threadIdx.x;
    const int lane  = tid & 31;
    const int w     = tid >> 5;
    const int ml    = lane & 15;
    const int kb    = (lane < 16) ? 0 : 8;

    const __bf16* qbase = qs + (size_t)bh * NN * HD;
    const __bf16* sbase = stateT + (size_t)bh * HD * HD;
    const float*  ks    = ksum + bh * HD;
    const int rowbase   = mtile * 64 + w * 16;

    __shared__ float den_s[4][16];
    if (lane < 16) {
        const __bf16* qr = qbase + (size_t)(rowbase + lane) * HD;
        float s = 0.f;
        for (int d0 = 0; d0 < HD; d0 += 8) {
            union { uint4 u; __bf16 hh[8]; } t;
            t.u = *(const uint4*)(qr + d0);
            #pragma unroll
            for (int i = 0; i < 8; ++i) s += (float)t.hh[i] * ks[d0 + i];
        }
        den_s[w][lane] = s + EPS;
    }
    __syncthreads();

    v8f acc[4] = {};
    for (int kk = 0; kk < HD; kk += 32) {
        Frag a;
        const __bf16* ar = qbase + (size_t)(rowbase + ml) * HD + kk + kb;
        a.u[0] = *(const uint4*)ar;
        a.u[1] = *(const uint4*)(ar + 16);
        #pragma unroll
        for (int j = 0; j < 4; ++j) {
            Frag bfr;
            const __bf16* br = sbase + (size_t)(j * 16 + ml) * HD + kk + kb;
            bfr.u[0] = *(const uint4*)br;
            bfr.u[1] = *(const uint4*)(br + 16);
            acc[j] = __builtin_amdgcn_wmma_f32_16x16x32_bf16(
                false, a.v, false, bfr.v, (short)0, acc[j], false, false);
        }
    }

    #pragma unroll
    for (int j = 0; j < 4; ++j)
        #pragma unroll
        for (int e = 0; e < 8; ++e) {
            int m   = (lane < 16) ? e : e + 8;
            int n   = rowbase + m;
            int col = j * 16 + ml;
            float val = acc[j][e] / den_s[w][m];
            attn[((size_t)b * NN + n) * CC + h * HD + col] = (__bf16)val;
        }
}

// ------------------------------------------------------------------
// GEMM2: out = attn(16384x1024) * Wout + b_out  (fp32 out)
// ------------------------------------------------------------------
__global__ __launch_bounds__(256) void gemm_out_kernel(const __bf16* __restrict__ A,
                                                       const __bf16* __restrict__ WT,
                                                       const float*  __restrict__ bias,
                                                       float* __restrict__ out) {
    const int tid    = threadIdx.x;
    const int lane   = tid & 31;
    const int wid    = tid >> 5;
    const int wave_m = wid >> 2;
    const int wave_n = wid & 3;
    const int row0   = blockIdx.y * 128;
    const int col0   = blockIdx.x * 256;

    __shared__ __bf16 As[2][128 * LDA];
    __shared__ __bf16 Bs[2][256 * LDA];

    v8f acc[4][4] = {};
    const int ml = lane & 15;
    const int kb = (lane < 16) ? 0 : 8;
    const int NT = CC / 32;

    stage_tiles(A, WT, row0, col0, 0, As[0], Bs[0], tid);

    for (int it = 0; it < NT; ++it) {
        int cur = it & 1;
        if (it + 1 < NT) {
            stage_tiles(A, WT, row0, col0, (it + 1) * 32, As[1 - cur], Bs[1 - cur], tid);
            WAIT_ASYNC(6);
        } else {
            WAIT_ASYNC(0);
        }
        __syncthreads();

        const __bf16* Ab = As[cur];
        const __bf16* Bb = Bs[cur];

        Frag b[4];
        #pragma unroll
        for (int j = 0; j < 4; ++j) {
            int n = wave_n * 64 + j * 16 + ml;
            b[j].u[0] = *(const uint4*)&Bb[n * LDA + kb];
            b[j].u[1] = *(const uint4*)&Bb[n * LDA + kb + 16];
        }
        #pragma unroll
        for (int i = 0; i < 4; ++i) {
            Frag a;
            int m = wave_m * 64 + i * 16 + ml;
            a.u[0] = *(const uint4*)&Ab[m * LDA + kb];
            a.u[1] = *(const uint4*)&Ab[m * LDA + kb + 16];
            #pragma unroll
            for (int j = 0; j < 4; ++j)
                acc[i][j] = __builtin_amdgcn_wmma_f32_16x16x32_bf16(
                    false, a.v, false, b[j].v, (short)0, acc[i][j], false, false);
        }
        __syncthreads();
    }

    #pragma unroll
    for (int i = 0; i < 4; ++i)
        #pragma unroll
        for (int j = 0; j < 4; ++j)
            #pragma unroll
            for (int e = 0; e < 8; ++e) {
                int mloc = wave_m * 64 + i * 16 + ((lane < 16) ? e : e + 8);
                int nloc = wave_n * 64 + j * 16 + ml;
                int r = row0 + mloc;
                int f = col0 + nloc;
                out[(size_t)r * CC + f] = acc[i][j][e] + bias[f];
            }
}

// ------------------------------------------------------------------
extern "C" void kernel_launch(void* const* d_in, const int* in_sizes, int n_in,
                              void* d_out, int out_size, void* d_ws, size_t ws_size,
                              hipStream_t stream) {
    (void)in_sizes; (void)n_in; (void)out_size; (void)ws_size;
    const float* x    = (const float*)d_in[0];
    const float* Wqkv = (const float*)d_in[1];
    const float* bqkv = (const float*)d_in[2];
    const float* Wout = (const float*)d_in[3];
    const float* bout = (const float*)d_in[4];
    float* out = (float*)d_out;

    char* ws = (char*)d_ws;
    size_t off = 0;
    auto alloc = [&](size_t bytes) -> void* {
        void* p = ws + off;
        off += (bytes + 255) & ~(size_t)255;
        return p;
    };

    const size_t ROWS = (size_t)BB * NN;                 // 16384
    __bf16* xb     = (__bf16*)alloc(ROWS * CC * 2);
    __bf16* wqT    = (__bf16*)alloc((size_t)FF * CC * 2);   // Wqkv^T (3072 x 1024)
    __bf16* woT    = (__bf16*)alloc((size_t)CC * CC * 2);   // Wout^T (1024 x 1024)
    __bf16* qs     = (__bf16*)alloc(ROWS * CC * 2);      // (b,h,n,d)
    __bf16* kT     = (__bf16*)alloc(ROWS * CC * 2);      // (b,h,d,n)
    __bf16* vT     = (__bf16*)alloc(ROWS * CC * 2);      // (b,h,d,n)
    __bf16* stateT = (__bf16*)alloc((size_t)BB * HH * HD * HD * 2);
    float*  ksum   = (float*)alloc((size_t)BB * HH * HD * 4);
    __bf16* attn   = (__bf16*)alloc(ROWS * CC * 2);      // (b,n,c)

    int nx  = (int)(ROWS * CC);
    int nwq = CC * FF;
    int nwo = CC * CC;
    cvt_bf16_kernel  <<<(nx  + 255) / 256, 256, 0, stream>>>(x, xb, nx);
    cvt_bf16_t_kernel<<<(nwq + 255) / 256, 256, 0, stream>>>(Wqkv, wqT, CC, FF);
    cvt_bf16_t_kernel<<<(nwo + 255) / 256, 256, 0, stream>>>(Wout, woT, CC, CC);

    gemm_qkv_kernel<<<dim3(FF / 256, (int)(ROWS / 128)), 256, 0, stream>>>(
        xb, wqT, bqkv, qs, kT, vT);

    state_kernel<<<BB * HH, 128, 0, stream>>>(kT, vT, stateT, ksum);

    attn_kernel<<<dim3(NN / 64, BB * HH), 128, 0, stream>>>(qs, stateT, ksum, attn);

    gemm_out_kernel<<<dim3(CC / 256, (int)(ROWS / 128)), 256, 0, stream>>>(
        attn, woT, bout, out);
}
